// QueryDecoder_46239617909125
// MI455X (gfx1250) — matrix operand order repaired and checked by reading, not compiled, not run on hardware
//
#include <hip/hip_runtime.h>
#include <hip/hip_bf16.h>

// ---------------------------------------------------------------------------
// CDNA5 (gfx1250) QueryDecoder forward pass.
// All matmuls use v_wmma_f32_16x16x32_bf16 (wave32 WMMA). Operands are staged
// through LDS as bf16 in *fragment-major* order so every WMMA fragment is a
// contiguous 32-byte LDS load (2 x ds_load_b128).
//
// Fragment layouts (CDNA5 ISA):
//   A (16x32 bf16): lane(half,r): row=r, element e -> K = e + (e<8?0:8) + half*8
//     -> not used here: A rows staged unpermuted, element e -> K = e + half*16
//   B (32x16 bf16): col=r; this code reads element e as K = e + half*16 (A too)
//     -> stage tile with K innermost (contiguous) per column
//   C/D (16x16 f32): element j -> row = half*8 + j, col = r
// ---------------------------------------------------------------------------

typedef __attribute__((ext_vector_type(16))) __bf16 v16bf;
typedef __attribute__((ext_vector_type(8)))  __bf16 v8bf;
typedef __attribute__((ext_vector_type(2)))  __bf16 v2bf;
typedef __attribute__((ext_vector_type(8)))  float  v8f;

__device__ __forceinline__ v8f v8f_zero() {
  v8f r;
#pragma unroll
  for (int i = 0; i < 8; i++) r[i] = 0.0f;
  return r;
}

__device__ __forceinline__ v8f wmma_bf16(v16bf a, v16bf b, v8f c) {
  return __builtin_amdgcn_wmma_f32_16x16x32_bf16(false, a, false, b, (short)0, c,
                                                 false, false);
}

// Contiguous 32B fragment load as two 16B LDS loads.
__device__ __forceinline__ v16bf ld_frag(const __bf16* p) {
  v8bf lo = *(const v8bf*)p;
  v8bf hi = *(const v8bf*)(p + 8);
  return __builtin_shufflevector(lo, hi, 0, 1, 2, 3, 4, 5, 6, 7,
                                 8, 9, 10, 11, 12, 13, 14, 15);
}

__device__ __forceinline__ v8bf cvt8(float4 a, float4 b) {
  v8bf o;
  o[0] = (__bf16)a.x; o[1] = (__bf16)a.y; o[2] = (__bf16)a.z; o[3] = (__bf16)a.w;
  o[4] = (__bf16)b.x; o[5] = (__bf16)b.y; o[6] = (__bf16)b.z; o[7] = (__bf16)b.w;
  return o;
}

__device__ __forceinline__ v8bf v8bf_zero() {
  v8bf o;
#pragma unroll
  for (int i = 0; i < 8; i++) o[i] = (__bf16)0.0f;
  return o;
}

#define LDP 40  // padded LDS row stride (bf16 elems): keeps 16B alignment, breaks bank stride

// ---------------------------------------------------------------------------
// GEMM:  C[M,N] = A[M,K] @ W[K,N] (+bias) (+accumulate) (+relu)
// Block tile 128x64, 4 waves; each wave owns 32 rows (2 A-frags) and reuses the
// 4 B-frags -> 8 WMMAs per wave per K-step (K step 32, K multiple of 32).
// flags: bit0 relu, bit1 accumulate
// ---------------------------------------------------------------------------
__global__ __launch_bounds__(128)
void gemm_bf16_k(const float* __restrict__ A, const float* __restrict__ W,
                 const float* __restrict__ bias, float* __restrict__ C,
                 int M, int K, int N, int flags) {
  __shared__ __bf16 lA[128][LDP];  // fragment-major (natural K order)
  __shared__ __bf16 lBt[64][LDP];  // [n][k], k contiguous
  int tid  = threadIdx.x;
  int wv   = tid >> 5;
  int lane = tid & 31;
  int half = lane >> 4;
  int r    = lane & 15;
  int m0 = blockIdx.y * 128;
  int n0 = blockIdx.x * 64;

  v8f acc[2][4];
#pragma unroll
  for (int a = 0; a < 2; a++)
#pragma unroll
    for (int t = 0; t < 4; t++) acc[a][t] = v8f_zero();

  for (int kb = 0; kb < K; kb += 32) {
    // stage A: one full row (32 k) per thread; k-groups kept in natural order
    {
      int row = tid;  // 0..127
      int gm = m0 + row;
#pragma unroll
      for (int kg = 0; kg < 4; kg++) {
        int dg = kg;
        v8bf o = v8bf_zero();
        if (gm < M) {
          const float4* p = (const float4*)(A + (long)gm * K + kb + kg * 8);
          o = cvt8(p[0], p[1]);
        }
        *(v8bf*)&lA[row][dg * 8] = o;
      }
    }
    // stage W transposed: lBt[n][k]; coalesced reads across n, packed b32 stores
#pragma unroll
    for (int i = 0; i < 8; i++) {
      int idx = tid + 128 * i;       // 0..1023
      int n = idx & 63;
      int kp = idx >> 6;             // 0..15
      int gn = n0 + n;
      float w0 = 0.0f, w1 = 0.0f;
      if (gn < N) {
        w0 = W[(long)(kb + 2 * kp) * N + gn];
        w1 = W[(long)(kb + 2 * kp + 1) * N + gn];
      }
      v2bf pk;
      pk[0] = (__bf16)w0;
      pk[1] = (__bf16)w1;
      *(v2bf*)&lBt[n][2 * kp] = pk;
    }
    __syncthreads();

    v16bf af0 = ld_frag(&lA[wv * 32 + r][half * 16]);
    v16bf af1 = ld_frag(&lA[wv * 32 + 16 + r][half * 16]);
#pragma unroll
    for (int t = 0; t < 4; t++) {
      v16bf bfq = ld_frag(&lBt[t * 16 + r][half * 16]);
      acc[0][t] = wmma_bf16(af0, bfq, acc[0][t]);
      acc[1][t] = wmma_bf16(af1, bfq, acc[1][t]);
    }
    __syncthreads();
  }

#pragma unroll
  for (int a = 0; a < 2; a++) {
#pragma unroll
    for (int t = 0; t < 4; t++) {
#pragma unroll
      for (int j = 0; j < 8; j++) {
        int gm = m0 + wv * 32 + a * 16 + half * 8 + j;
        int gn = n0 + t * 16 + r;
        if (gm < M && gn < N) {
          float v = acc[a][t][j];
          if (bias) v += bias[gn];
          if (flags & 2) v += C[(long)gm * N + gn];
          if (flags & 1) v = fmaxf(v, 0.0f);
          C[(long)gm * N + gn] = v;
        }
      }
    }
  }
}

// ---------------------------------------------------------------------------
// Batched NT GEMM:  C[b][M,Nn] = A[b][M,K] @ B[b][Nn,K]^T   (pm / pm1)
// B rows are K-contiguous => natural layout is already fragment-major for B.
// Block tile 128x64 like gemm_bf16_k.
// ---------------------------------------------------------------------------
__global__ __launch_bounds__(128)
void gemm_nt_k(const float* __restrict__ A, const float* __restrict__ Bm,
               float* __restrict__ C, int M, int Nn, int K,
               long sA, long sB, long sC) {
  int b = blockIdx.z;
  const float* Ab = A + (long)b * sA;
  const float* Bb = Bm + (long)b * sB;
  float* Cb = C + (long)b * sC;

  __shared__ __bf16 lA[128][LDP];
  __shared__ __bf16 lBt[64][LDP];  // [n][k] natural
  int tid  = threadIdx.x;
  int wv   = tid >> 5;
  int lane = tid & 31;
  int half = lane >> 4;
  int r    = lane & 15;
  int m0 = blockIdx.y * 128;
  int n0 = blockIdx.x * 64;

  v8f acc[2][4];
#pragma unroll
  for (int a = 0; a < 2; a++)
#pragma unroll
    for (int t = 0; t < 4; t++) acc[a][t] = v8f_zero();

  for (int kb = 0; kb < K; kb += 32) {
    {
      int row = tid;
      int gm = m0 + row;
#pragma unroll
      for (int kg = 0; kg < 4; kg++) {
        int dg = kg;
        v8bf o = v8bf_zero();
        if (gm < M) {
          const float4* p = (const float4*)(Ab + (long)gm * K + kb + kg * 8);
          o = cvt8(p[0], p[1]);
        }
        *(v8bf*)&lA[row][dg * 8] = o;
      }
    }
#pragma unroll
    for (int gI = 0; gI < 2; gI++) {
      int g = tid * 2 + gI;
      int row = g >> 2;  // n index
      int kg = g & 3;
      int gn = n0 + row;
      v8bf o = v8bf_zero();
      if (gn < Nn) {
        const float4* p = (const float4*)(Bb + (long)gn * K + kb + kg * 8);
        o = cvt8(p[0], p[1]);
      }
      *(v8bf*)&lBt[row][kg * 8] = o;
    }
    __syncthreads();

    v16bf af0 = ld_frag(&lA[wv * 32 + r][half * 16]);
    v16bf af1 = ld_frag(&lA[wv * 32 + 16 + r][half * 16]);
#pragma unroll
    for (int t = 0; t < 4; t++) {
      v16bf bfq = ld_frag(&lBt[t * 16 + r][half * 16]);
      acc[0][t] = wmma_bf16(af0, bfq, acc[0][t]);
      acc[1][t] = wmma_bf16(af1, bfq, acc[1][t]);
    }
    __syncthreads();
  }

#pragma unroll
  for (int a = 0; a < 2; a++) {
#pragma unroll
    for (int t = 0; t < 4; t++) {
#pragma unroll
      for (int j = 0; j < 8; j++) {
        int gm = m0 + wv * 32 + a * 16 + half * 8 + j;
        int gn = n0 + t * 16 + r;
        if (gm < M && gn < Nn) Cb[(long)gm * Nn + gn] = acc[a][t][j];
      }
    }
  }
}

// ---------------------------------------------------------------------------
// Flash multi-head attention. Q/K/V: [B*L, 256], head h at channel h*32.
// hd=32 == one bf16 WMMA K-depth. mask uint8 (True==disallowed -> -1e9) or null.
// 128 threads = 4 waves = 64 query rows; key chunks of 32; online softmax with
// 16-lane shfl_xor row reductions; P written to LDS directly in A-frag layout.
// ---------------------------------------------------------------------------
__global__ __launch_bounds__(128)
void attn_flash_k(const float* __restrict__ Qp, const float* __restrict__ Kp,
                  const float* __restrict__ Vp, const unsigned char* __restrict__ mask,
                  float* __restrict__ Out, int Lq, int Lk, float scale) {
  int b = blockIdx.z, h = blockIdx.y;
  int q0 = blockIdx.x * 64;
  int tid  = threadIdx.x;
  int wv   = tid >> 5;
  int lane = tid & 31;
  int half = lane >> 4;
  int r    = lane & 15;

  __shared__ __bf16 lQ[64][LDP];       // fragment-major (natural d order)
  __shared__ __bf16 lK[32][LDP];       // [key][d] natural (d contiguous)
  __shared__ __bf16 lVt[32][LDP];      // [d][key] transposed (key contiguous)
  __shared__ __bf16 lP[4][16][LDP];    // per-wave P tile, A-frag layout

  // stage Q (natural K groups)
#pragma unroll
  for (int gI = 0; gI < 2; gI++) {
    int g = tid * 2 + gI;
    int row = g >> 2;
    int kg = g & 3;
    int dg = kg;
    int gq = q0 + row;
    v8bf o = v8bf_zero();
    if (gq < Lq) {
      const float4* p =
          (const float4*)(Qp + ((long)b * Lq + gq) * 256 + h * 32 + kg * 8);
      o = cvt8(p[0], p[1]);
    }
    *(v8bf*)&lQ[row][dg * 8] = o;
  }
  __syncthreads();
  v16bf aq = ld_frag(&lQ[wv * 16 + r][half * 16]);

  float mrow[8], lrow[8];
#pragma unroll
  for (int j = 0; j < 8; j++) { mrow[j] = -3.0e38f; lrow[j] = 0.0f; }
  v8f o0 = v8f_zero(), o1 = v8f_zero();

  int posa = r;                    // P store slot for key r
  int posb = r + 16;               // P store slot for key 16+r

  for (int kb = 0; kb < Lk; kb += 32) {
    __syncthreads();
    // stage K natural: 32 rows x 4 groups -> one group per thread
    {
      int row = tid >> 2;
      int kg = tid & 3;
      int gk = kb + row;
      v8bf o = v8bf_zero();
      if (gk < Lk) {
        const float4* p =
            (const float4*)(Kp + ((long)b * Lk + gk) * 256 + h * 32 + kg * 8);
        o = cvt8(p[0], p[1]);
      }
      *(v8bf*)&lK[row][kg * 8] = o;
    }
    // stage V transposed: lVt[d][key]
#pragma unroll
    for (int i = 0; i < 4; i++) {
      int idx = tid + 128 * i;     // 0..511
      int d = idx & 31;
      int kp = idx >> 5;           // 0..15
      int gk0 = kb + 2 * kp;
      float v0 = 0.0f, v1 = 0.0f;
      if (gk0 < Lk)     v0 = Vp[((long)b * Lk + gk0) * 256 + h * 32 + d];
      if (gk0 + 1 < Lk) v1 = Vp[((long)b * Lk + gk0 + 1) * 256 + h * 32 + d];
      v2bf pk;
      pk[0] = (__bf16)v0;
      pk[1] = (__bf16)v1;
      *(v2bf*)&lVt[d][2 * kp] = pk;
    }
    __syncthreads();

    v16bf bk0 = ld_frag(&lK[r][half * 16]);        // col=key r,    K=d
    v16bf bk1 = ld_frag(&lK[16 + r][half * 16]);   // col=key 16+r, K=d
    v16bf bv0 = ld_frag(&lVt[r][half * 16]);       // col=d r,      K=key
    v16bf bv1 = ld_frag(&lVt[16 + r][half * 16]);  // col=d 16+r,   K=key

    v8f s0 = wmma_bf16(aq, bk0, v8f_zero());
    v8f s1 = wmma_bf16(aq, bk1, v8f_zero());

#pragma unroll
    for (int j = 0; j < 8; j++) {
      int gq = q0 + wv * 16 + half * 8 + j;
      int k0i = kb + r, k1i = kb + 16 + r;
      float v0 = s0[j] * scale, v1 = s1[j] * scale;
      if (k0i >= Lk) v0 = -3.0e38f;
      else if (mask && gq < Lq && mask[((long)b * Lq + gq) * (long)Lk + k0i]) v0 = -1.0e9f;
      if (k1i >= Lk) v1 = -3.0e38f;
      else if (mask && gq < Lq && mask[((long)b * Lq + gq) * (long)Lk + k1i]) v1 = -1.0e9f;

      float cm = fmaxf(v0, v1);
#pragma unroll
      for (int off = 1; off < 16; off <<= 1) cm = fmaxf(cm, __shfl_xor(cm, off, 32));
      float mnew = fmaxf(mrow[j], cm);
      float p0 = __expf(v0 - mnew);
      float p1 = __expf(v1 - mnew);
      float alpha = __expf(mrow[j] - mnew);
      float cs = p0 + p1;
#pragma unroll
      for (int off = 1; off < 16; off <<= 1) cs += __shfl_xor(cs, off, 32);
      lrow[j] = lrow[j] * alpha + cs;
      mrow[j] = mnew;
      o0[j] *= alpha;
      o1[j] *= alpha;
      lP[wv][half * 8 + j][posa] = (__bf16)p0;
      lP[wv][half * 8 + j][posb] = (__bf16)p1;
    }
    v16bf ap = ld_frag(&lP[wv][r][half * 16]);
    o0 = wmma_bf16(ap, bv0, o0);
    o1 = wmma_bf16(ap, bv1, o1);
  }

#pragma unroll
  for (int j = 0; j < 8; j++) {
    int gq = q0 + wv * 16 + half * 8 + j;
    if (gq < Lq) {
      float inv = 1.0f / lrow[j];
      long base = ((long)b * Lq + gq) * 256 + h * 32;
      Out[base + r] = o0[j] * inv;
      Out[base + 16 + r] = o1[j] * inv;
    }
  }
}

// ---------------------------------------------------------------------------
// Elementwise / small kernels
// ---------------------------------------------------------------------------
__global__ __launch_bounds__(256)
void ln_k(const float* __restrict__ X, const float* __restrict__ Res,
          const float* __restrict__ g, const float* __restrict__ be,
          float* __restrict__ O, int relu) {
  int row = blockIdx.x;
  int c = threadIdx.x;
  long base = (long)row * 256 + c;
  float v = X[base];
  if (Res) v += Res[base];
  __shared__ float red[256];
  red[c] = v;
  __syncthreads();
  for (int s = 128; s > 0; s >>= 1) { if (c < s) red[c] += red[c + s]; __syncthreads(); }
  float mean = red[0] * (1.0f / 256.0f);
  __syncthreads();
  float d = v - mean;
  red[c] = d * d;
  __syncthreads();
  for (int s = 128; s > 0; s >>= 1) { if (c < s) red[c] += red[c + s]; __syncthreads(); }
  float var = red[0] * (1.0f / 256.0f);
  float y = g[c] * d * rsqrtf(var + 1e-5f) + be[c];
  if (relu) y = fmaxf(y, 0.0f);
  O[base] = y;
}

__global__ __launch_bounds__(256)
void bn_relu_k(float* __restrict__ Y, const float* __restrict__ g,
               const float* __restrict__ be, const float* __restrict__ mean,
               const float* __restrict__ var, long total) {
  long i = (long)blockIdx.x * 256 + threadIdx.x;
  if (i >= total) return;
  int c = (int)(i & 255);
  float v = Y[i];
  v = g[c] * (v - mean[c]) * rsqrtf(var[c] + 1e-5f) + be[c];
  Y[i] = fmaxf(v, 0.0f);
}

__global__ __launch_bounds__(256)
void add_k(const float* __restrict__ A, const float* __restrict__ B,
           float* __restrict__ O, long n) {
  long i = (long)blockIdx.x * 256 + threadIdx.x;
  if (i < n) O[i] = A[i] + B[i];
}

__global__ __launch_bounds__(256)
void copy_k(const float* __restrict__ A, float* __restrict__ O, long n) {
  long i = (long)blockIdx.x * 256 + threadIdx.x;
  if (i < n) O[i] = A[i];
}

__global__ __launch_bounds__(256)
void bcast_query_k(const float* __restrict__ Q, float* __restrict__ O0,
                   float* __restrict__ O1, int per, long total) {
  long i = (long)blockIdx.x * 256 + threadIdx.x;
  if (i < total) {
    float v = Q[i % per];
    O0[i] = v;
    O1[i] = v;
  }
}

// am = (sigmoid(pm) < 0.5) & ~all(row)   with sigmoid(pm)<0.5 <=> pm<0
__global__ __launch_bounds__(256)
void fixmask_k(const float* __restrict__ pm, unsigned char* __restrict__ am, int Lk) {
  int row = blockIdx.x;
  __shared__ int allv;
  if (threadIdx.x == 0) allv = 1;
  __syncthreads();
  int localAll = 1;
  for (int k = threadIdx.x; k < Lk; k += 256)
    if (!(pm[(long)row * Lk + k] < 0.0f)) localAll = 0;
  if (!localAll) atomicAnd(&allv, 0);
  __syncthreads();
  int at = allv;
  for (int k = threadIdx.x; k < Lk; k += 256)
    am[(long)row * Lk + k] =
        (unsigned char)((pm[(long)row * Lk + k] < 0.0f) && !at);
}

// Naive GEMM for tiny N (19 / 1)
__global__ __launch_bounds__(256)
void smallgemm_k(const float* __restrict__ A, const float* __restrict__ W,
                 const float* __restrict__ bias, float* __restrict__ C,
                 int M, int K, int N) {
  int idx = blockIdx.x * 256 + threadIdx.x;
  if (idx >= M * N) return;
  int m = idx / N, n = idx % N;
  float s = bias ? bias[n] : 0.0f;
  for (int k = 0; k < K; k++) s += A[(long)m * K + k] * W[(long)k * N + n];
  C[idx] = s;
}

// ---------------------------------------------------------------------------
// Host orchestration
// ---------------------------------------------------------------------------
extern "C" void kernel_launch(void* const* d_in, const int* in_sizes, int n_in,
                              void* d_out, int out_size, void* d_ws, size_t ws_size,
                              hipStream_t stream) {
  (void)n_in; (void)out_size;
  const float* x          = (const float*)d_in[0];   // [B*N, 32]
  const float* query_feat = (const float*)d_in[1];   // [B*NQ, 64]
  int Bq = in_sizes[2] - 1;                          // 8
  int MN = in_sizes[0] / 32;                         // 65536
  int Nn = MN / Bq;                                  // 8192
  int MQ = in_sizes[1] / 64;                         // 800
  int NQ = MQ / Bq;                                  // 100

  // ---- params (jax pytree flatten: dict keys sorted alphabetically) ----
  int pi = 3;
  auto nx = [&]() { return (const float*)d_in[pi++]; };
  const float *fc_b = nx(), *fc_w = nx();                              // fc: b, w[512,256]
  const float *ip_b = nx(), *ip_be = nx(), *ip_g = nx(), *ip_w = nx(); // input_proj
  struct LayerP {
    const float *c_bi, *c_bo, *c_wi, *c_wo;
    const float *f_b1, *f_b2, *f_be, *f_g, *f_w1, *f_w2;
    const float *s_be, *s_bi, *s_bo, *s_g, *s_wi, *s_wo;
  } Lp[6];
  for (int l = 0; l < 6; l++) {
    Lp[l].c_bi = nx(); Lp[l].c_bo = nx(); Lp[l].c_wi = nx(); Lp[l].c_wo = nx();
    Lp[l].f_b1 = nx(); Lp[l].f_b2 = nx(); Lp[l].f_be = nx(); Lp[l].f_g = nx();
    Lp[l].f_w1 = nx(); Lp[l].f_w2 = nx();
    Lp[l].s_be = nx(); Lp[l].s_bi = nx(); Lp[l].s_bo = nx(); Lp[l].s_g = nx();
    Lp[l].s_wi = nx(); Lp[l].s_wo = nx();
  }
  struct MT { const float *b, *be, *g, *mean, *var, *w; } mt[3];
  for (int i = 0; i < 3; i++) {
    mt[i].b = nx(); mt[i].be = nx(); mt[i].g = nx();
    mt[i].mean = nx(); mt[i].var = nx(); mt[i].w = nx();
  }
  const float *mt3_b = nx(), *mt3_w = nx();
  const float *oc_b1 = nx(), *oc_b2 = nx(), *oc_w1 = nx(), *oc_w2 = nx();
  const float *on_be = nx(), *on_g = nx();
  const float *os_b1 = nx(), *os_b2 = nx(), *os_w1 = nx(), *os_w2 = nx();
  const float *qparam = nx();
  const float *ql_b = nx(), *ql_be = nx(), *ql_g = nx(), *ql_w = nx();
  const float *xm_b1 = nx(), *xm_b2 = nx(), *xm_w1 = nx(), *xm_w2 = nx();

  // ---- workspace carve ----
  char* wsp = (char*)d_ws;
  auto carve = [&](size_t bytes) {
    char* p = wsp;
    wsp += (bytes + 255) & ~(size_t)255;
    return (void*)p;
  };
  float* inst   = (float*)carve((size_t)MN * 256 * 4);
  float* mf     = (float*)carve((size_t)MN * 256 * 4);
  float* Kb     = (float*)carve((size_t)MN * 256 * 4);
  float* Vb     = (float*)carve((size_t)MN * 256 * 4);
  float* qf     = (float*)carve((size_t)MQ * 256 * 4);
  float* h0     = (float*)carve((size_t)MQ * 256 * 4);
  float* h1     = (float*)carve((size_t)MQ * 256 * 4);
  float* query  = (float*)carve((size_t)MQ * 256 * 4);
  float* query1 = (float*)carve((size_t)MQ * 256 * 4);
  float* qn     = (float*)carve((size_t)MQ * 256 * 4);
  float* t0     = (float*)carve((size_t)MQ * 256 * 4);
  float* t1     = (float*)carve((size_t)MQ * 256 * 4);
  float* t2     = (float*)carve((size_t)MQ * 256 * 4);
  float* KQ1    = (float*)carve((size_t)MQ * 256 * 4);
  float* VQ1    = (float*)carve((size_t)MQ * 256 * 4);
  float* ffnh   = (float*)carve((size_t)MQ * 1024 * 4);
  float* pmbuf  = (float*)carve((size_t)MQ * Nn * 4);
  float* pm1buf = (float*)carve((size_t)MQ * NQ * 4);
  unsigned char* am  = (unsigned char*)carve((size_t)MQ * Nn);
  unsigned char* am1 = (unsigned char*)carve((size_t)MQ * NQ);
  if ((size_t)(wsp - (char*)d_ws) > ws_size) return;  // insufficient scratch

  const float scale = 0.17677669529663687f;  // 1/sqrt(32)

  auto gemm = [&](const float* A, const float* W, const float* bias, float* C,
                  int M, int Kd, int Nd, int flags) {
    dim3 g((Nd + 63) / 64, (M + 127) / 128, 1);
    gemm_bf16_k<<<g, dim3(128), 0, stream>>>(A, W, bias, C, M, Kd, Nd, flags);
  };
  auto gemmnt = [&](const float* A, const float* Bm, float* C, int M, int Nd, int Kd,
                    long sA, long sB, long sC) {
    dim3 g((Nd + 63) / 64, (M + 127) / 128, Bq);
    gemm_nt_k<<<g, dim3(128), 0, stream>>>(A, Bm, C, M, Nd, Kd, sA, sB, sC);
  };
  auto ln = [&](const float* X, const float* Res, const float* g_, const float* be,
                float* O, int M, int relu) {
    ln_k<<<dim3(M), dim3(256), 0, stream>>>(X, Res, g_, be, O, relu);
  };
  auto attn = [&](const float* Qp, const float* Kp, const float* Vp,
                  const unsigned char* m, float* O, int Lq, int Lk) {
    dim3 g((Lq + 63) / 64, 8, Bq);
    attn_flash_k<<<g, dim3(128), 0, stream>>>(Qp, Kp, Vp, m, O, Lq, Lk, scale);
  };
  auto addv = [&](const float* A, const float* B, float* O, long n) {
    add_k<<<dim3((unsigned)((n + 255) / 256)), dim3(256), 0, stream>>>(A, B, O, n);
  };

  // ---- 1) inst = relu(LN(x @ ip_w + ip_b)) ----
  gemm(x, ip_w, ip_b, Kb, MN, 32, 256, 0);
  ln(Kb, nullptr, ip_g, ip_be, inst, MN, 1);

  // ---- 2) mf = relu(x @ xm_w1 + b1) @ xm_w2 + b2 ----
  gemm(x, xm_w1, xm_b1, Vb, MN, 32, 256, 1);
  gemm(Vb, xm_w2, xm_b2, mf, MN, 256, 256, 0);

  // ---- 3) h = relu(LN(query_feat @ ql_w + ql_b)) ----
  gemm(query_feat, ql_w, ql_b, h0, MQ, 64, 256, 0);
  ln(h0, nullptr, ql_g, ql_be, h0, MQ, 1);

  // ---- 4) mask tower -> qf ----
  float* hc = h0;
  float* hn = h1;
  for (int i = 0; i < 3; i++) {
    gemm(hc, mt[i].w, mt[i].b, hn, MQ, 256, 256, 0);
    bn_relu_k<<<dim3(MQ), dim3(256), 0, stream>>>(hn, mt[i].g, mt[i].be,
                                                  mt[i].mean, mt[i].var,
                                                  (long)MQ * 256);
    float* tsw = hc; hc = hn; hn = tsw;
  }
  gemm(hc, mt3_w, mt3_b, qf, MQ, 256, 256, 0);

  // ---- 5) query / query1 init ----
  bcast_query_k<<<dim3((MQ * 256 + 255) / 256), dim3(256), 0, stream>>>(
      qparam, query, query1, NQ * 256, (long)MQ * 256);

  // ---- 6) prediction head (masks only, used by next layer) ----
  auto predhead_masks = [&]() {
    ln(query, nullptr, on_g, on_be, qn, MQ, 0);
    gemmnt(qn, mf, pmbuf, NQ, Nn, 256, (long)NQ * 256, (long)Nn * 256, (long)NQ * Nn);
    gemmnt(qn, qf, pm1buf, NQ, NQ, 256, (long)NQ * 256, (long)NQ * 256, (long)NQ * NQ);
    fixmask_k<<<dim3(MQ), dim3(256), 0, stream>>>(pmbuf, am, Nn);
    fixmask_k<<<dim3(MQ), dim3(256), 0, stream>>>(pm1buf, am1, NQ);
  };
  predhead_masks();

  // ---- 7) decoder layers ----
  for (int l = 0; l < 6; l++) {
    const LayerP& P = Lp[l];
    const float *cwq = P.c_wi, *cwk = P.c_wi + 65536, *cwv = P.c_wi + 131072;
    const float *cbq = P.c_bi, *cbk = P.c_bi + 256, *cbv = P.c_bi + 512;
    const float *swq = P.s_wi, *swk = P.s_wi + 65536, *swv = P.s_wi + 131072;
    const float *sbq = P.s_bi, *sbk = P.s_bi + 256, *sbv = P.s_bi + 512;

    // (a) query1 cross-attn vs qf (mask am1) + residual
    gemm(query1, cwq, cbq, t0, MQ, 256, 256, 0);
    gemm(qf, cwk, cbk, KQ1, MQ, 256, 256, 0);
    gemm(qf, cwv, cbv, VQ1, MQ, 256, 256, 0);
    attn(t0, KQ1, VQ1, am1, t1, NQ, NQ);
    gemm(t1, P.c_wo, P.c_bo, t2, MQ, 256, 256, 0);
    addv(t2, query1, query1, (long)MQ * 256);

    // (b) query cross-attn vs inst (mask am) + residual
    gemm(query, cwq, cbq, t0, MQ, 256, 256, 0);
    gemm(inst, cwk, cbk, Kb, MN, 256, 256, 0);
    gemm(inst, cwv, cbv, Vb, MN, 256, 256, 0);
    attn(t0, Kb, Vb, am, t1, NQ, Nn);
    gemm(t1, P.c_wo, P.c_bo, t2, MQ, 256, 256, 0);
    addv(t2, query, query, (long)MQ * 256);

    // (c) self-attn query1 -> LN(attn + residual)
    gemm(query1, swq, sbq, t0, MQ, 256, 256, 0);
    gemm(query1, swk, sbk, KQ1, MQ, 256, 256, 0);
    gemm(query1, swv, sbv, VQ1, MQ, 256, 256, 0);
    attn(t0, KQ1, VQ1, nullptr, t1, NQ, NQ);
    gemm(t1, P.s_wo, P.s_bo, t2, MQ, 256, 256, 0);
    ln(t2, query1, P.s_g, P.s_be, query1, MQ, 0);

    // (d) self-attn query
    gemm(query, swq, sbq, t0, MQ, 256, 256, 0);
    gemm(query, swk, sbk, KQ1, MQ, 256, 256, 0);
    gemm(query, swv, sbv, VQ1, MQ, 256, 256, 0);
    attn(t0, KQ1, VQ1, nullptr, t1, NQ, NQ);
    gemm(t1, P.s_wo, P.s_bo, t2, MQ, 256, 256, 0);
    ln(t2, query, P.s_g, P.s_be, query, MQ, 0);

    // (e) FFN query1
    gemm(query1, P.f_w1, P.f_b1, ffnh, MQ, 256, 1024, 1);
    gemm(ffnh, P.f_w2, P.f_b2, t2, MQ, 1024, 256, 0);
    ln(t2, query1, P.f_g, P.f_be, query1, MQ, 0);

    // (f) FFN query
    gemm(query, P.f_w1, P.f_b1, ffnh, MQ, 256, 1024, 1);
    gemm(ffnh, P.f_w2, P.f_b2, t2, MQ, 1024, 256, 0);
    ln(t2, query, P.f_g, P.f_be, query, MQ, 0);

    // (g) query = concat([query, query1]) @ fc_w + fc_b  (split-K)
    gemm(query, fc_w, fc_b, t0, MQ, 256, 256, 0);
    gemm(query1, fc_w + 256 * 256, nullptr, t0, MQ, 256, 256, 2);
    copy_k<<<dim3((MQ * 256 + 255) / 256), dim3(256), 0, stream>>>(
        t0, query, (long)MQ * 256);

    // (h) prediction head
    if (l < 5) {
      predhead_masks();
    } else {
      float* outp = (float*)d_out;
      ln(query, nullptr, on_g, on_be, qn, MQ, 0);
      gemm(qn, oc_w1, oc_b1, t0, MQ, 256, 256, 1);
      smallgemm_k<<<dim3((MQ * 19 + 255) / 256), dim3(256), 0, stream>>>(
          t0, oc_w2, oc_b2, outp, MQ, 256, 19);                       // labels
      gemm(qn, os_w1, os_b1, t0, MQ, 256, 256, 1);
      smallgemm_k<<<dim3((MQ * 1 + 255) / 256), dim3(256), 0, stream>>>(
          t0, os_w2, os_b2, outp + (long)MQ * 19, MQ, 256, 1);        // scores
      gemmnt(qn, mf, outp + (long)MQ * 19 + MQ, NQ, Nn, 256,
             (long)NQ * 256, (long)Nn * 256, (long)NQ * Nn);          // pm
    }
  }
}
